// CrossAttentionViT_20048907338134
// MI455X (gfx1250) — matrix operand order repaired and checked by reading, not compiled, not run on hardware
//
#include <hip/hip_runtime.h>
#include <hip/hip_bf16.h>

// ---------------------------------------------------------------------------
// CrossAttentionViT forward, MI455X (gfx1250), bf16 WMMA pipeline.
//
// Algebraic simplifications (exact, not approximations):
//  * cross-attention has 1 query x 1 key  -> softmax == 1 -> attn_out =
//    ((kv@Wv^T+bv)@Wo^T+bo); ca_Wq/ca_Wk/ca_bq/ca_bk are dead.
//  * only pair[:,0] feeds the classifier -> token-1 Q proj and token-1 output
//    proj of the self-attention are dead.
//
// GEMMs run on v_wmma_f32_16x16x32_bf16. K-tile staging uses the CDNA5 async
// global->LDS path (GLOBAL_LOAD_ASYNC_TO_LDS_B128 + s_wait_asynccnt) with
// double-buffered LDS when the toolchain exposes the builtins; otherwise the
// synchronous staging path is used.
// ---------------------------------------------------------------------------

typedef __bf16 bf16;
typedef __attribute__((ext_vector_type(16))) __bf16 v16bf;
typedef __attribute__((ext_vector_type(8)))  float  v8f;

#define BB   16384
#define SDIM 2048
#define FDIM 1024
#define DD   1024

#define BM 128
#define BN 128
#define BK 32
#define LDSS 40   // LDS row stride in halves (32 data + 8 pad), 80B rows

#if defined(__HIP_DEVICE_COMPILE__) &&                                     \
    __has_builtin(__builtin_amdgcn_global_load_async_to_lds_b128) &&       \
    __has_builtin(__builtin_amdgcn_s_wait_asynccnt)
#define HAS_ASYNC_LDS 1
// exact parameter type per hipcc diagnostic: int __attribute__((vector_size(16)))
typedef int v4i_vs __attribute__((vector_size(16)));
typedef __attribute__((address_space(1))) v4i_vs* gasync_p;
typedef __attribute__((address_space(3))) v4i_vs* lasync_p;
#else
#define HAS_ASYNC_LDS 0
#endif

static __device__ __forceinline__ bf16 f2bf(float f) {
  unsigned int u = __builtin_bit_cast(unsigned int, f);
  u = (u + 0x7FFFu + ((u >> 16) & 1u)) >> 16;          // round-to-nearest-even
  unsigned short s = (unsigned short)u;
  return __builtin_bit_cast(bf16, s);
}
static __device__ __forceinline__ float bf2f(bf16 h) {
  unsigned short s = __builtin_bit_cast(unsigned short, h);
  unsigned int u = ((unsigned int)s) << 16;
  return __builtin_bit_cast(float, u);
}
static __device__ __forceinline__ unsigned short bfbits(float f) {
  unsigned int u = __builtin_bit_cast(unsigned int, f);
  u = (u + 0x7FFFu + ((u >> 16) & 1u)) >> 16;
  return (unsigned short)u;
}

// ---------------------------------------------------------------------------
// f32 -> bf16 bulk conversion (4-wide; all sizes here are multiples of 4)
// ---------------------------------------------------------------------------
__global__ __launch_bounds__(256) void k_cvt_bf16(
    const float* __restrict__ src, bf16* __restrict__ dst, long n4) {
  long i = (long)blockIdx.x * blockDim.x + threadIdx.x;
  if (i >= n4) return;
  float4 f = ((const float4*)src)[i];
  ushort4 o;
  o.x = bfbits(f.x); o.y = bfbits(f.y); o.z = bfbits(f.z); o.w = bfbits(f.w);
  ((ushort4*)dst)[i] = o;
}

__global__ __launch_bounds__(256) void k_combine_bias(
    const float* __restrict__ b, const float* __restrict__ pos,
    float* __restrict__ out, int n) {
  int i = blockIdx.x * blockDim.x + threadIdx.x;
  if (i < n) out[i] = b[i] + pos[i];
}

// ---------------------------------------------------------------------------
// Tiled bf16 WMMA GEMM:  C[M,N](bf16) = act( A[M,K](bf16) @ W[N,K]^T + bias )
// 256 threads = 8 waves; wave grid 2x4; 64x32 per wave = 4x2 WMMA tiles.
// act: 0 = identity, 1 = exact GELU.
// ---------------------------------------------------------------------------
__global__ __launch_bounds__(256) void k_gemm_bf16(
    const bf16* __restrict__ A, const bf16* __restrict__ W,
    const float* __restrict__ bias, bf16* __restrict__ C,
    int M, int N, int K, int act) {
  __shared__ bf16 As[2][BM * LDSS];
  __shared__ bf16 Ws[2][BN * LDSS];

  const int tid  = threadIdx.x;
  const int lane = tid & 31;
  const int wave = tid >> 5;
  const int wm   = wave & 1;   // 0..1  (64-row slabs)
  const int wn   = wave >> 1;  // 0..3  (32-col slabs)
  const int half = lane >> 4;  // 0/1
  const int l16  = lane & 15;

  const long blockM = (long)blockIdx.y * BM;
  const long blockN = (long)blockIdx.x * BN;

  v8f acc[4][2];
#pragma unroll
  for (int i = 0; i < 4; ++i)
#pragma unroll
    for (int j = 0; j < 2; ++j)
#pragma unroll
      for (int r = 0; r < 8; ++r) acc[i][j][r] = 0.0f;

  // staging: each thread moves 16 halves (32B) of A and of W per K-tile
  const int srow = tid >> 1;          // 0..127
  const int scol = (tid & 1) * 16;    // 0 / 16
  const int ktiles = K / BK;

  // one K-step of WMMA math out of LDS buffer `buf`
  auto compute = [&](int buf) {
    v16bf afrag[4], bfrag[2];
    // A 16x32 bf16 fragment layout (ISA 7.12.2): lane<16 row M=l16 holds
    // K {0..7,16..23}; lane>=16 same row holds K {8..15,24..31}.
#pragma unroll
    for (int i = 0; i < 4; ++i) {
      const bf16* p  = &As[buf][(wm * 64 + i * 16 + l16) * LDSS];
      const bf16* p0 = p + half * 8;
      const bf16* p1 = p + 16 + half * 8;
      v16bf f;
#pragma unroll
      for (int e = 0; e < 8; ++e) { f[e] = p0[e]; f[8 + e] = p1[e]; }
      afrag[i] = f;
    }
    // B 32x16 fragment: column n of B == row n of W; lane<16 holds
    // K {0..7,16..23}; lane>=16 holds K {8..15,24..31} (same as A).
#pragma unroll
    for (int j = 0; j < 2; ++j) {
      const bf16* p = &Ws[buf][(wn * 32 + j * 16 + l16) * LDSS] + half * 8;
      v16bf f;
#pragma unroll
      for (int e = 0; e < 16; ++e) f[e] = p[e < 8 ? e : e + 8];
      bfrag[j] = f;
    }
#pragma unroll
    for (int i = 0; i < 4; ++i)
#pragma unroll
      for (int j = 0; j < 2; ++j)
        acc[i][j] = __builtin_amdgcn_wmma_f32_16x16x32_bf16(
            false, afrag[i], false, bfrag[j], (short)0, acc[i][j],
            false, false);
  };

#if HAS_ASYNC_LDS
  // CDNA5 async global->LDS staging, double buffered. Each thread issues
  // 4x GLOBAL_LOAD_ASYNC_TO_LDS_B128 per K-tile (ASYNCcnt-tracked, in-order
  // completion). s_wait_asynccnt(4) => the *previous* tile's 4 ops landed;
  // the barrier publishes them across waves.
  auto stage_async = [&](int kt, int buf) {
    const long kb = (long)kt * BK;
    const bf16* ag = A + (blockM + srow) * (long)K + kb + scol;
    const bf16* wg = W + (blockN + srow) * (long)K + kb + scol;
    bf16* al = &As[buf][srow * LDSS + scol];
    bf16* wl = &Ws[buf][srow * LDSS + scol];
    __builtin_amdgcn_global_load_async_to_lds_b128(
        (gasync_p)(ag), (lasync_p)(al), 0, 0);
    __builtin_amdgcn_global_load_async_to_lds_b128(
        (gasync_p)(ag + 8), (lasync_p)(al + 8), 0, 0);
    __builtin_amdgcn_global_load_async_to_lds_b128(
        (gasync_p)(wg), (lasync_p)(wl), 0, 0);
    __builtin_amdgcn_global_load_async_to_lds_b128(
        (gasync_p)(wg + 8), (lasync_p)(wl + 8), 0, 0);
  };

  stage_async(0, 0);
  for (int kt = 0; kt < ktiles; ++kt) {
    const int buf = kt & 1;
    if (kt + 1 < ktiles) {
      stage_async(kt + 1, buf ^ 1);          // prefetch next tile into other buf
      __builtin_amdgcn_s_wait_asynccnt(4);   // this tile's 4 ops complete
    } else {
      __builtin_amdgcn_s_wait_asynccnt(0);
    }
    __syncthreads();                         // publish LDS across waves
    compute(buf);
    __syncthreads();                         // everyone done with buf
  }
#else
  // Synchronous fallback: global->VGPR->LDS with software prefetch.
  for (int kt = 0; kt < ktiles; ++kt) {
    const int buf = kt & 1;
    const long kb = (long)kt * BK;
    const uint4* ag = (const uint4*)(A + (blockM + srow) * (long)K + kb + scol);
    const uint4* wg = (const uint4*)(W + (blockN + srow) * (long)K + kb + scol);
    uint4 a0 = ag[0], a1 = ag[1];
    uint4 w0 = wg[0], w1 = wg[1];
    if (kt + 1 < ktiles) {            // global_prefetch_b8 next K-tile
      __builtin_prefetch(A + (blockM + srow) * (long)K + kb + BK + scol, 0, 1);
      __builtin_prefetch(W + (blockN + srow) * (long)K + kb + BK + scol, 0, 1);
    }
    uint4* asd = (uint4*)&As[buf][srow * LDSS + scol];
    asd[0] = a0; asd[1] = a1;
    uint4* wsd = (uint4*)&Ws[buf][srow * LDSS + scol];
    wsd[0] = w0; wsd[1] = w1;
    __syncthreads();
    compute(buf);
    __syncthreads();
  }
#endif

  // epilogue: C/D layout (ISA 7.12.2): VGPR r holds M = r + 8*(lane>=16),
  // N = lane&15.
  const int rowHi = half * 8;
#pragma unroll
  for (int i = 0; i < 4; ++i) {
    const long rowBase = blockM + wm * 64 + i * 16 + rowHi;
#pragma unroll
    for (int j = 0; j < 2; ++j) {
      const long col = blockN + wn * 32 + j * 16 + l16;
      const float bv = bias ? bias[col] : 0.0f;
#pragma unroll
      for (int r = 0; r < 8; ++r) {
        float x = acc[i][j][r] + bv;
        if (act == 1) x = 0.5f * x * (1.0f + erff(x * 0.70710678118654752f));
        C[(rowBase + r) * (long)N + col] = f2bf(x);
      }
    }
  }
}

// ---------------------------------------------------------------------------
// out = LayerNorm(X + R) * g + b      (rows of length 1024)
// ---------------------------------------------------------------------------
__global__ __launch_bounds__(256) void k_add_ln(
    const bf16* __restrict__ X, const bf16* __restrict__ R,
    const float* __restrict__ g, const float* __restrict__ b,
    bf16* __restrict__ O) {
  __shared__ float sA[8], sB[8];
  const long row = blockIdx.x;
  const int tid = threadIdx.x, lane = tid & 31, wave = tid >> 5;
  const bf16* x = X + row * DD;
  const bf16* r = R + row * DD;
  float v[4], s = 0.f, s2 = 0.f;
#pragma unroll
  for (int e = 0; e < 4; ++e) {
    int idx = tid * 4 + e;
    float t = bf2f(x[idx]) + bf2f(r[idx]);
    v[e] = t; s += t; s2 += t * t;
  }
#pragma unroll
  for (int off = 16; off > 0; off >>= 1) {
    s  += __shfl_xor(s,  off, 32);
    s2 += __shfl_xor(s2, off, 32);
  }
  if (lane == 0) { sA[wave] = s; sB[wave] = s2; }
  __syncthreads();
  float ts = 0.f, ts2 = 0.f;
#pragma unroll
  for (int w = 0; w < 8; ++w) { ts += sA[w]; ts2 += sB[w]; }
  const float mu = ts * (1.0f / DD);
  const float var = ts2 * (1.0f / DD) - mu * mu;
  const float rstd = rsqrtf(var + 1e-5f);
  bf16* o = O + row * DD;
#pragma unroll
  for (int e = 0; e < 4; ++e) {
    int idx = tid * 4 + e;
    o[idx] = f2bf((v[e] - mu) * rstd * g[idx] + b[idx]);
  }
}

// ---------------------------------------------------------------------------
// 2-token self-attention, row 0 only:
//   o0[h] = softmax((q0.k0, q0.k1)/sqrt(128)) . [v0[h]; v1[h]]
// one wave per head (8 heads, d=128 -> 4 elems/lane)
// ---------------------------------------------------------------------------
__global__ __launch_bounds__(256) void k_sa_attn(
    const bf16* __restrict__ Q0, const bf16* __restrict__ K0,
    const bf16* __restrict__ K1, const bf16* __restrict__ V0,
    const bf16* __restrict__ V1, bf16* __restrict__ O0) {
  const long row = blockIdx.x;
  const int tid = threadIdx.x;
  const int h = tid >> 5, lane = tid & 31;
  const long base = row * DD + (long)h * 128;
  float s00 = 0.f, s01 = 0.f;
#pragma unroll
  for (int e = 0; e < 4; ++e) {
    int idx = lane * 4 + e;
    float q = bf2f(Q0[base + idx]);
    s00 += q * bf2f(K0[base + idx]);
    s01 += q * bf2f(K1[base + idx]);
  }
#pragma unroll
  for (int off = 16; off > 0; off >>= 1) {
    s00 += __shfl_xor(s00, off, 32);
    s01 += __shfl_xor(s01, off, 32);
  }
  const float scale = 0.08838834764831845f;  // 1/sqrt(128)
  s00 *= scale; s01 *= scale;
  const float m = fmaxf(s00, s01);
  const float e0 = __expf(s00 - m), e1 = __expf(s01 - m);
  const float inv = 1.0f / (e0 + e1);
  const float a0 = e0 * inv, a1 = e1 * inv;
#pragma unroll
  for (int e = 0; e < 4; ++e) {
    int idx = lane * 4 + e;
    O0[base + idx] =
        f2bf(a0 * bf2f(V0[base + idx]) + a1 * bf2f(V1[base + idx]));
  }
}

// ---------------------------------------------------------------------------
// out[b] = x3[b,:] . Wc + bc   (one wave per row)
// ---------------------------------------------------------------------------
__global__ __launch_bounds__(256) void k_cls_head(
    const bf16* __restrict__ X, const bf16* __restrict__ Wc,
    const float* __restrict__ bc, float* __restrict__ out) {
  const int tid = threadIdx.x, lane = tid & 31, wave = tid >> 5;
  const long row = (long)blockIdx.x * 8 + wave;
  const bf16* x = X + row * DD;
  float s = 0.f;
#pragma unroll
  for (int e = 0; e < 32; ++e) {
    int idx = e * 32 + lane;
    s += bf2f(x[idx]) * bf2f(Wc[idx]);
  }
#pragma unroll
  for (int off = 16; off > 0; off >>= 1) s += __shfl_xor(s, off, 32);
  if (lane == 0) out[row] = s + bc[0];
}

// ---------------------------------------------------------------------------
extern "C" void kernel_launch(void* const* d_in, const int* in_sizes, int n_in,
                              void* d_out, int out_size, void* d_ws,
                              size_t ws_size, hipStream_t stream) {
  (void)in_sizes; (void)n_in; (void)out_size; (void)ws_size;

  char* base = (char*)d_ws;
  size_t off = 0;
  auto take = [&](size_t elems, size_t esz) -> size_t {
    size_t r = off;
    off += (elems * esz + 255) & ~(size_t)255;
    return r;
  };

  // bf16 weight arena
  size_t o_wps = take((size_t)DD * SDIM, 2);
  size_t o_wpf = take((size_t)DD * FDIM, 2);
  size_t o_cav = take((size_t)DD * DD, 2);
  size_t o_cao = take((size_t)DD * DD, 2);
  size_t o_m1  = take((size_t)2 * DD * DD, 2);
  size_t o_m2  = take((size_t)DD * 2 * DD, 2);
  size_t o_saq = take((size_t)DD * DD, 2);
  size_t o_sak = take((size_t)DD * DD, 2);
  size_t o_sav = take((size_t)DD * DD, 2);
  size_t o_sao = take((size_t)DD * DD, 2);
  size_t o_wc  = take((size_t)DD, 2);
  // f32 combined biases
  size_t o_bs = take(DD, 4);
  size_t o_bf = take(DD, 4);
  // bf16 activations
  size_t o_insp = take((size_t)BB * SDIM, 2);
  size_t o_infr = take((size_t)BB * FDIM, 2);
  size_t o_q0  = take((size_t)BB * DD, 2);
  size_t o_kv  = take((size_t)BB * DD, 2);
  size_t o_vca = take((size_t)BB * DD, 2);
  size_t o_att = take((size_t)BB * DD, 2);
  size_t o_x1  = take((size_t)BB * DD, 2);
  size_t o_h1  = take((size_t)BB * 2 * DD, 2);
  size_t o_h2  = take((size_t)BB * DD, 2);
  size_t o_x2  = take((size_t)BB * DD, 2);
  size_t o_qt0 = take((size_t)BB * DD, 2);
  size_t o_kt0 = take((size_t)BB * DD, 2);
  size_t o_kt1 = take((size_t)BB * DD, 2);
  size_t o_vt0 = take((size_t)BB * DD, 2);
  size_t o_vt1 = take((size_t)BB * DD, 2);
  size_t o_o0  = take((size_t)BB * DD, 2);
  size_t o_po0 = take((size_t)BB * DD, 2);
  size_t o_x3  = take((size_t)BB * DD, 2);

  auto BF = [&](size_t o) { return (bf16*)(base + o); };
  auto F32 = [&](size_t o) { return (float*)(base + o); };
  auto IN = [&](int i) { return (const float*)d_in[i]; };

  // ---- conversions (weights + big inputs) ----
  auto cvt = [&](const float* s, bf16* d, long n) {
    long n4 = n / 4;
    k_cvt_bf16<<<dim3((unsigned)((n4 + 255) / 256)), dim3(256), 0, stream>>>(
        s, d, n4);
  };
  cvt(IN(2),  BF(o_wps), (long)DD * SDIM);   // Wps
  cvt(IN(4),  BF(o_wpf), (long)DD * FDIM);   // Wpf
  cvt(IN(10), BF(o_cav), (long)DD * DD);     // ca_Wv
  cvt(IN(14), BF(o_cao), (long)DD * DD);     // ca_Wo
  cvt(IN(18), BF(o_m1),  (long)2 * DD * DD); // mlp_W1
  cvt(IN(20), BF(o_m2),  (long)DD * 2 * DD); // mlp_W2
  cvt(IN(24), BF(o_saq), (long)DD * DD);     // sa_Wq
  cvt(IN(25), BF(o_sak), (long)DD * DD);     // sa_Wk
  cvt(IN(26), BF(o_sav), (long)DD * DD);     // sa_Wv
  cvt(IN(30), BF(o_sao), (long)DD * DD);     // sa_Wo
  cvt(IN(34), BF(o_wc),  (long)DD);          // Wc
  cvt(IN(0),  BF(o_insp), (long)BB * SDIM);  // spatial_feat
  cvt(IN(1),  BF(o_infr), (long)BB * FDIM);  // freq_feat

  k_combine_bias<<<dim3(4), dim3(256), 0, stream>>>(IN(3), IN(6), F32(o_bs), DD);
  k_combine_bias<<<dim3(4), dim3(256), 0, stream>>>(IN(5), IN(7), F32(o_bf), DD);

  // ---- GEMM helper ----
  auto G = [&](const bf16* A, const bf16* W, const float* bias, bf16* C,
               int M, int N, int K, int act) {
    dim3 grid((unsigned)(N / BN), (unsigned)(M / BM));
    k_gemm_bf16<<<grid, dim3(256), 0, stream>>>(A, W, bias, C, M, N, K, act);
  };

  // 1-2: input projections (+pos folded into bias)
  G(BF(o_insp), BF(o_wps), F32(o_bs), BF(o_q0), BB, DD, SDIM, 0);
  G(BF(o_infr), BF(o_wpf), F32(o_bf), BF(o_kv), BB, DD, FDIM, 0);
  // 3-4: cross-attention collapsed (softmax over 1 key == 1)
  G(BF(o_kv),  BF(o_cav), IN(13), BF(o_vca), BB, DD, DD, 0);
  G(BF(o_vca), BF(o_cao), IN(15), BF(o_att), BB, DD, DD, 0);
  // 5: x1 = LN(q0 + attn)
  k_add_ln<<<dim3(BB), dim3(256), 0, stream>>>(BF(o_q0), BF(o_att), IN(16),
                                               IN(17), BF(o_x1));
  // 6-7: MLP (exact GELU fused into GEMM 1)
  G(BF(o_x1), BF(o_m1), IN(19), BF(o_h1), BB, 2 * DD, DD, 1);
  G(BF(o_h1), BF(o_m2), IN(21), BF(o_h2), BB, DD, 2 * DD, 0);
  // 8: x2 = LN(x1 + h)
  k_add_ln<<<dim3(BB), dim3(256), 0, stream>>>(BF(o_x1), BF(o_h2), IN(22),
                                               IN(23), BF(o_x2));
  // 9: 2-token self-attention (token-0 output only)
  G(BF(o_x2), BF(o_saq), IN(27), BF(o_qt0), BB, DD, DD, 0);
  G(BF(o_x2), BF(o_sak), IN(28), BF(o_kt0), BB, DD, DD, 0);
  G(BF(o_kv), BF(o_sak), IN(28), BF(o_kt1), BB, DD, DD, 0);
  G(BF(o_x2), BF(o_sav), IN(29), BF(o_vt0), BB, DD, DD, 0);
  G(BF(o_kv), BF(o_sav), IN(29), BF(o_vt1), BB, DD, DD, 0);
  k_sa_attn<<<dim3(BB), dim3(256), 0, stream>>>(BF(o_qt0), BF(o_kt0),
                                                BF(o_kt1), BF(o_vt0),
                                                BF(o_vt1), BF(o_o0));
  G(BF(o_o0), BF(o_sao), IN(31), BF(o_po0), BB, DD, DD, 0);
  // x3 = LN(x2 + pair_out0)
  k_add_ln<<<dim3(BB), dim3(256), 0, stream>>>(BF(o_x2), BF(o_po0), IN(32),
                                               IN(33), BF(o_x3));
  // 10: classifier head
  k_cls_head<<<dim3(BB / 8), dim3(256), 0, stream>>>(BF(o_x3), BF(o_wc),
                                                     IN(35), (float*)d_out);
}
